// GRU_Model2_47837345743487
// MI455X (gfx1250) — compile-verified, matrix operand order checked by reading, not run on hardware
//
#include <hip/hip_runtime.h>
#include <hip/hip_bf16.h>
#include <stdint.h>

// ---------------- types ----------------
typedef __attribute__((ext_vector_type(16))) __bf16 v16bf;
typedef __attribute__((ext_vector_type(8)))  float  v8f;
typedef __attribute__((ext_vector_type(4)))  unsigned int tdmv4u;
typedef __attribute__((ext_vector_type(8)))  int          tdmv8i;
typedef __attribute__((ext_vector_type(4)))  int          tdmv4i;

union ABu { v16bf v; uint4 q[2]; unsigned short s[16]; };

// ---------------- problem dims ----------------
#define B_TOT   2048
#define T_STEPS 256
#define F_IN    32
#define H_DIM   256
#define N3H     768
#define BM      16                  // batch rows per workgroup
#define NWG     (B_TOT / BM)        // 128 workgroups
#define NWAVES  8
#define NTHR    (NWAVES * 32)       // 256 threads
#define KCH     8                   // 256 / 32 K-chunks
#define KCH_LDS 4                   // K-chunks of rec_kernel cached in LDS

// ---------------- workspace layout (bytes) ----------------
#define XBF_BYTES   (128ull * 256ull * 1024ull)   // [btile][t][lane][16 bf16]
#define BHHALF_B    (48u * 4u * 1024u)            // 48 ntiles x 4 kchunks x 1KB
#define BX_BYTES    (48u * 1024u)
#define WS_XBF      0ull
#define WS_BHLO     (WS_XBF + XBF_BYTES)
#define WS_BHHI     (WS_BHLO + BHHALF_B)
#define WS_BX       (WS_BHHI + BHHALF_B)

// ---------------- LDS layout (bytes) ----------------
#define L_BX    0u
#define L_BHLO  (L_BX + 48u * 1024u)              // 49152
#define L_HBF   (L_BHLO + 48u * 4u * 1024u)       // 245760
#define HROW    272                                // padded row stride (bf16 elems)
#define SMEM_BYTES (L_HBF + 16u * HROW * 2u)      // 254464

// ---------------- helpers ----------------
__device__ __forceinline__ unsigned short f32_to_bf16(float f) {
    unsigned u = __float_as_uint(f);
    u += 0x7FFFu + ((u >> 16) & 1u);              // round-to-nearest-even
    return (unsigned short)(u >> 16);
}
__device__ __forceinline__ float sigmoidf_(float x) {
    return 1.0f / (1.0f + __expf(-x));
}
// 16-bit A/B fragment element mapping: element e (0..15), lane-group hi (0/1) -> k
__host__ __device__ __forceinline__ int kmap(int e, int hi) {
    return ((e >= 8) ? 16 : 0) + (e & 7) + hi * 8;
}
__device__ __forceinline__ v8f wmma_bf16(v16bf a, v16bf b, v8f c) {
    return __builtin_amdgcn_wmma_f32_16x16x32_bf16(false, a, false, b, (short)0, c,
                                                   false, false);
}
__device__ __forceinline__ v16bf ldsB(const char* smem, unsigned off) {
    ABu b;
    b.q[0] = *(const uint4*)(smem + off);
    b.q[1] = *(const uint4*)(smem + off + 16);
    return b.v;
}
__device__ __forceinline__ v16bf glbB(const unsigned short* p) {
    ABu b;
    b.q[0] = *(const uint4*)(p);
    b.q[1] = *(const uint4*)(p + 8);
    return b.v;
}

#if __has_builtin(__builtin_amdgcn_tensor_load_to_lds)
// 1-D TDM copy: `bytes` contiguous bytes global -> LDS (bytes % 4 == 0, elems <= 65535)
__device__ __forceinline__ void tdm_copy_1d(const void* gsrc, unsigned lds_off,
                                            unsigned bytes) {
    unsigned long long ga = (unsigned long long)(uintptr_t)gsrc;
    unsigned elems = bytes >> 2;   // 4-byte elements
    tdmv4u g0 = { 1u,                                   // count=1, user descriptor
                  lds_off,                              // lds_addr
                  (unsigned)(ga & 0xFFFFFFFFu),         // global_addr[31:0]
                  (unsigned)((ga >> 32) & 0x1FFFFFFu) | 0x80000000u }; // addr hi | type=2
    tdmv8i g1 = { (int)(2u << 16),                              // data_size=4B
                  (int)((elems & 0xFFFFu) << 16),               // tensor_dim0 lo16
                  (int)(((elems >> 16) & 0xFFFFu) | (1u << 16)),// dim0 hi | tensor_dim1=1
                  (int)((elems & 0xFFFFu) << 16),               // tile_dim0 = elems
                  (int)1,                                        // tile_dim1 = 1
                  (int)elems,                                    // dim0_stride lo32
                  (int)(((elems >> 16) & 0xFFFFu) | ((elems & 0xFFFFu) << 16)),
                  (int)((elems >> 16) & 0xFFFFu) };
    tdmv4i g2 = { 1, 0, 0, 0 };
    tdmv4i g3 = { 0, 0, 0, 0 };
    tdmv8i g4 = { 0, 0, 0, 0, 0, 0, 0, 0 };     // extended group (6-arg toolchain form)
    __builtin_amdgcn_tensor_load_to_lds(g0, g1, g2, g3, g4, 0);
}
#endif

// ---------------- prep: x fp32 -> bf16 A-fragment layout ----------------
__global__ void prep_x(const float* __restrict__ x, unsigned short* __restrict__ xbf) {
    unsigned o = blockIdx.x * blockDim.x + threadIdx.x;   // 16,777,216 total
    int e = o & 15, lane = (o >> 4) & 31, t = (o >> 9) & 255, btile = o >> 17;
    int m = lane & 15, hi = lane >> 4;
    int f = kmap(e, hi);
    int b = btile * 16 + m;
    xbf[o] = f32_to_bf16(x[((size_t)b * T_STEPS + t) * F_IN + f]);
}

// ---------------- prep: rec_kernel / kernel -> bf16 B-tile layout ----------------
__global__ void prep_B(const float* __restrict__ rk, const float* __restrict__ kn,
                       unsigned short* __restrict__ bh, unsigned short* __restrict__ bx) {
    unsigned o = blockIdx.x * blockDim.x + threadIdx.x;   // 221,184 total
    int e = o & 15, lane = (o >> 4) & 31;
    int n = lane & 15, hi = lane >> 4;
    int kk = kmap(e, hi);
    if (o < 48u * 8u * 512u) {
        int kc   = (o >> 9) & 7;
        int tile = o >> 12;
        int col  = tile * 16 + n;
        int krow = kc * 32 + kk;
        unsigned short v = f32_to_bf16(rk[(size_t)krow * N3H + col]);
        unsigned base = (kc < 4) ? ((unsigned)(tile * 4 + kc) * 512u)
                                 : (98304u + (unsigned)(tile * 4 + kc - 4) * 512u);
        bh[base + ((unsigned)lane << 4) + e] = v;
    } else if (o < 48u * 8u * 512u + 48u * 512u) {
        unsigned oo = o - 48u * 8u * 512u;
        int tile = oo >> 9;
        int col  = tile * 16 + n;
        bx[(unsigned)tile * 512u + ((unsigned)lane << 4) + e] =
            f32_to_bf16(kn[(size_t)kk * N3H + col]);
    }
}

// ---------------- main persistent GRU kernel ----------------
__global__ __launch_bounds__(NTHR) void gru_main(
    const unsigned short* __restrict__ xbf,
    const unsigned short* __restrict__ bhlo_g,
    const unsigned short* __restrict__ bhhi_g,
    const unsigned short* __restrict__ bx_g,
    const float* __restrict__ bias,
    const float* __restrict__ gamma, const float* __restrict__ beta,
    const float* __restrict__ mean,  const float* __restrict__ var,
    const float* __restrict__ dw,    const float* __restrict__ db,
    float* __restrict__ out)
{
    extern __shared__ char smem[];
    const int tid  = threadIdx.x;
    const int w    = tid >> 5;
    const int lane = tid & 31;
    const int n    = lane & 15;
    const int hi   = lane >> 4;
    const int btile = blockIdx.x;

    unsigned short* hbf = (unsigned short*)(smem + L_HBF);

    // zero h buffer (h0 = 0)
    for (int i = tid; i < 16 * HROW / 2; i += NTHR) ((unsigned*)hbf)[i] = 0u;

    // stage kernel-tiles (48KB) + first 4 K-chunks of rec_kernel (192KB) into LDS
#if __has_builtin(__builtin_amdgcn_tensor_load_to_lds)
    if (w == 0) {
        tdm_copy_1d(bhlo_g, (unsigned)(uintptr_t)(smem + L_BHLO), BHHALF_B);
        tdm_copy_1d(bx_g,   (unsigned)(uintptr_t)(smem + L_BX),   BX_BYTES);
        __builtin_amdgcn_s_wait_tensorcnt(0);
    }
#else
    for (unsigned i = tid; i < (BHHALF_B >> 4); i += NTHR)
        ((uint4*)(smem + L_BHLO))[i] = ((const uint4*)bhlo_g)[i];
    for (unsigned i = tid; i < (BX_BYTES >> 4); i += NTHR)
        ((uint4*)(smem + L_BX))[i] = ((const uint4*)bx_g)[i];
#endif
    __syncthreads();

    // per-lane column constants (wave w owns column groups g=w and g=w+8)
    const int cols[2] = { 16 * w + n, 16 * (w + 8) + n };
    float bz[2], br[2], bxh[2], bih[2], scl[2], shf[2];
#pragma unroll
    for (int j = 0; j < 2; ++j) {
        int c = cols[j];
        bz[j]  = bias[c]       + bias[N3H + c];
        br[j]  = bias[256 + c] + bias[N3H + 256 + c];
        bxh[j] = bias[512 + c];
        bih[j] = bias[N3H + 512 + c];
        float rs  = rsqrtf(var[c] + 1e-3f);
        float gsc = rs * gamma[c];
        scl[j] = gsc * dw[c];
        shf[j] = (beta[c] - mean[c] * gsc) * dw[c];
    }

    v8f hstate0 = {}, hstate1 = {};
    const unsigned lb = (unsigned)lane * 32u;
    const unsigned short* xbase = xbf + (size_t)btile * T_STEPS * 512u + (unsigned)lane * 16u;
    const unsigned ahbase = (unsigned)n * HROW + (unsigned)hi * 8u;

    for (int t = 0; t < T_STEPS; ++t) {
        // A_x fragment (16x32 bf16, K = F = 32)
        ABu ax;
        const unsigned short* xp = xbase + (size_t)t * 512u;
        ax.q[0] = *(const uint4*)(xp);
        ax.q[1] = *(const uint4*)(xp + 8);

        // A_h fragments (16x32 bf16 per K-chunk) from LDS h buffer
        ABu ah[KCH];
#pragma unroll
        for (int kc = 0; kc < KCH; ++kc) {
            const unsigned short* hp = hbf + ahbase + kc * 32;
            ah[kc].q[0] = *(const uint4*)(hp);
            ah[kc].q[1] = *(const uint4*)(hp + 16);
        }
        __syncthreads();   // everyone done reading h(t)

#pragma unroll
        for (int j = 0; j < 2; ++j) {
            const int g  = (j == 0) ? w : (w + 8);
            const int tz = g, tr = 16 + g, th = 32 + g;
            v8f zero = {};
            // x-projection (biases added later in activation)
            v8f accZ  = wmma_bf16(ax.v, ldsB(smem, L_BX + (unsigned)tz * 1024u + lb), zero);
            v8f accR  = wmma_bf16(ax.v, ldsB(smem, L_BX + (unsigned)tr * 1024u + lb), zero);
            v8f accXH = wmma_bf16(ax.v, ldsB(smem, L_BX + (unsigned)th * 1024u + lb), zero);
            v8f accIH = zero;
#pragma unroll
            for (int kc = 0; kc < KCH; ++kc) {
                v16bf bzt, brt, bht;
                if (kc < KCH_LDS) {
                    bzt = ldsB(smem, L_BHLO + (unsigned)(tz * 4 + kc) * 1024u + lb);
                    brt = ldsB(smem, L_BHLO + (unsigned)(tr * 4 + kc) * 1024u + lb);
                    bht = ldsB(smem, L_BHLO + (unsigned)(th * 4 + kc) * 1024u + lb);
                } else {
                    bzt = glbB(bhhi_g + (unsigned)(tz * 4 + kc - 4) * 512u + (unsigned)lane * 16u);
                    brt = glbB(bhhi_g + (unsigned)(tr * 4 + kc - 4) * 512u + (unsigned)lane * 16u);
                    bht = glbB(bhhi_g + (unsigned)(th * 4 + kc - 4) * 512u + (unsigned)lane * 16u);
                }
                accZ  = wmma_bf16(ah[kc].v, bzt, accZ);
                accR  = wmma_bf16(ah[kc].v, brt, accR);
                accIH = wmma_bf16(ah[kc].v, bht, accIH);
            }
            // GRU gate math in D-layout (lane = column, vgpr index = row)
            v8f h = (j == 0) ? hstate0 : hstate1;
            v8f hn;
#pragma unroll
            for (int i = 0; i < 8; ++i) {
                float z  = sigmoidf_(accZ[i] + bz[j]);
                float r  = sigmoidf_(accR[i] + br[j]);
                float hh = accXH[i] + bxh[j] + r * (accIH[i] + bih[j]);
                hh = fmaxf(hh, 0.0f);
                hn[i] = z * h[i] + (1.0f - z) * hh;
            }
            if (j == 0) hstate0 = hn; else hstate1 = hn;
            // write new h (bf16) back for next step's A fragments
            const int c = cols[j];
#pragma unroll
            for (int i = 0; i < 8; ++i)
                hbf[(unsigned)(i + hi * 8) * HROW + c] = f32_to_bf16(hn[i]);
        }
        __syncthreads();   // h(t+1) fully written
    }

    // ---- BN + Dense(1): deterministic LDS reduction (reuse Bx region) ----
    __syncthreads();
    float* part = (float*)smem;   // [wave][row 16][col-slot 32]
#pragma unroll
    for (int j = 0; j < 2; ++j) {
        v8f h = (j == 0) ? hstate0 : hstate1;
#pragma unroll
        for (int i = 0; i < 8; ++i) {
            int m = i + hi * 8;
            part[(unsigned)(w * 16 + m) * 32u + (unsigned)(j * 16 + n)] =
                h[i] * scl[j] + shf[j];
        }
    }
    __syncthreads();
    if (tid < 16) {
        float s = 0.0f;
        for (int w2 = 0; w2 < NWAVES; ++w2)
            for (int c = 0; c < 32; ++c)
                s += part[(unsigned)(w2 * 16 + tid) * 32u + c];
        out[btile * 16 + tid] = s + db[0];
    }
}

// ---------------- host launcher ----------------
extern "C" void kernel_launch(void* const* d_in, const int* in_sizes, int n_in,
                              void* d_out, int out_size, void* d_ws, size_t ws_size,
                              hipStream_t stream) {
    (void)in_sizes; (void)n_in; (void)out_size; (void)ws_size;
    const float* x    = (const float*)d_in[0];
    const float* kern = (const float*)d_in[1];
    const float* rk   = (const float*)d_in[2];
    const float* bias = (const float*)d_in[3];
    const float* gmm  = (const float*)d_in[4];
    const float* bta  = (const float*)d_in[5];
    const float* mmn  = (const float*)d_in[6];
    const float* mvr  = (const float*)d_in[7];
    const float* dwp  = (const float*)d_in[8];
    const float* dbp  = (const float*)d_in[9];

    char* ws = (char*)d_ws;
    unsigned short* xbf  = (unsigned short*)(ws + WS_XBF);
    unsigned short* bhlo = (unsigned short*)(ws + WS_BHLO);
    unsigned short* bhhi = (unsigned short*)(ws + WS_BHHI);
    unsigned short* bx   = (unsigned short*)(ws + WS_BX);

    (void)hipFuncSetAttribute((const void*)gru_main,
                              hipFuncAttributeMaxDynamicSharedMemorySize, SMEM_BYTES);

    prep_x<<<65536, NTHR, 0, stream>>>(x, xbf);
    prep_B<<<864, NTHR, 0, stream>>>(rk, kern, bhlo, bx);
    gru_main<<<NWG, NTHR, SMEM_BYTES, stream>>>(xbf, bhlo, bhhi, bx, bias,
                                                gmm, bta, mmn, mvr, dwp, dbp,
                                                (float*)d_out);
}
